// Pct_67388036874863
// MI455X (gfx1250) — compile-verified, matrix operand order checked by reading, not run on hardware
//
#include <hip/hip_runtime.h>

// =====================================================================
// Point-Cloud-Transformer forward for MI455X (gfx1250, wave32).
//
// All GEMM-heavy math (local-op convs, SA projections, attention,
// head linears) runs through one LDS-staged WMMA kernel using
// v_wmma_f32_16x16x32_f16 (f16 operands, f32 accumulate).
// Irregular work (FPS, ball query, grouping, BN stats, softmax) uses
// plain scalar kernels.
//
// GEMM kernel: 64x64 C-tile / 4 waves, K staged 64-deep in LDS.
// Interior tiles use 16B global_load_b128 + ds_store_b128 (all f16
// leading dims are multiples of 8); edge tiles / K-tails use a
// branchless clamped-load + select path.
//
// Input flattening assumption: setup_inputs() dict order, nested params
// expanded depth-first in insertion order (72 tensors total, x first).
// =====================================================================

typedef __attribute__((ext_vector_type(16))) _Float16 v16h;
typedef __attribute__((ext_vector_type(8)))  _Float16 v8h;
typedef __attribute__((ext_vector_type(8)))  float    v8f;

#define GEMM_THREADS 128
#define TM 64
#define TN 64
#define TK 64
#define LDSK 72   // padded K stride in halves (144B rows -> 16B aligned frags)

static inline unsigned nblk(long long n) { return (unsigned)((n + 255) / 256); }

// ---------------------------------------------------------------------
// WMMA helpers
// ---------------------------------------------------------------------
__device__ inline v8f wmma_f32(v16h a, v16h b, v8f c) {
  return __builtin_amdgcn_wmma_f32_16x16x32_f16(
      /*neg_a=*/false, a, /*neg_b=*/false, b,
      /*c_mod=*/(short)0, c, /*reuse_a=*/false, /*reuse_b=*/false);
}

// Load one 16x32 f16 fragment from LDS (row-major, LDSK stride).
// Per ISA 16-bit A layout: lane L<16 holds row L, halves 0..7 -> k=kb..kb+7,
// halves 8..15 -> k=kb+16..kb+23 (kb = 0 for lanes<16, 8 otherwise).
__device__ inline v16h frag_ld(const _Float16* rowbase, int r16, int kb) {
  const _Float16* p = rowbase + r16 * LDSK + kb;
  union { v16h v; v8h h[2]; } u;
  u.h[0] = *(const v8h*)(p);
  u.h[1] = *(const v8h*)(p + 16);
  return u.v;
}

// ---------------------------------------------------------------------
// Batched NT GEMM:  C[M,N] = A[M,K] * B'[N,K]^T   (A,B' f16 row-major,
// K innermost for both; C f32). Batch via grid.z with element strides.
// Requires lda/ldb to be multiples of 8 (all call sites satisfy this).
// ---------------------------------------------------------------------
__global__ void __launch_bounds__(GEMM_THREADS)
k_gemm_nt(const _Float16* __restrict__ A, const _Float16* __restrict__ Bm,
          float* __restrict__ C,
          int M, int N, int K, int lda, int ldb, int ldc,
          long long sA, long long sB, long long sC)
{
  __shared__ _Float16 As[TM * LDSK];
  __shared__ _Float16 Bs[TN * LDSK];

  const int z = blockIdx.z;
  A  += (long long)z * sA;
  Bm += (long long)z * sB;
  C  += (long long)z * sC;

  const int m0 = blockIdx.y * TM;
  const int n0 = blockIdx.x * TN;

  const int tid  = threadIdx.x;
  const int lane = tid & 31;
  const int w    = tid >> 5;
  const int wm   = (w >> 1) * 32;
  const int wn   = (w & 1) * 32;

  v8f acc00 = {0,0,0,0,0,0,0,0};
  v8f acc01 = acc00, acc10 = acc00, acc11 = acc00;

  const int r16 = lane & 15;
  const int kb  = (lane >> 4) << 3;

  const bool inner = (m0 + TM <= M) && (n0 + TN <= N);

  // fast-path addressing: 2 threads per row, 4 x 16B chunks each
  const int frow = tid >> 1;               // 0..63
  const int fcb  = (tid & 1) * 4;          // chunk base (of 8 chunks/row)

  for (int k0 = 0; k0 < K; k0 += TK) {
    if (inner && (k0 + TK <= K)) {
      // -------- fully interior: vector panel loads --------
      const _Float16* ga = A  + (long long)(m0 + frow) * lda + k0 + fcb * 8;
      const _Float16* gb = Bm + (long long)(n0 + frow) * ldb + k0 + fcb * 8;
      _Float16* la = &As[frow * LDSK + fcb * 8];
      _Float16* lb = &Bs[frow * LDSK + fcb * 8];
      #pragma unroll
      for (int c = 0; c < 4; ++c) {
        v8h va = *(const v8h*)(ga + c * 8);
        v8h vb = *(const v8h*)(gb + c * 8);
        *(v8h*)(la + c * 8) = va;
        *(v8h*)(lb + c * 8) = vb;
      }
      if (k0 + 2 * TK <= K) {  // speculative prefetch of next panels
        __builtin_prefetch(ga + TK, 0, 1);
        __builtin_prefetch(gb + TK, 0, 1);
      }
    } else {
      // -------- edge / K-tail: branchless clamped loads --------
      #pragma unroll
      for (int t = 0; t < 16; ++t) {
        int d   = t * GEMM_THREADS + tid;   // 0..2047
        int row = d >> 5;                    // 0..63
        int cp  = d & 31;                    // half-pair index
        int k   = k0 + cp * 2;
        int kc  = k < K ? k : 0;
        int ra  = m0 + row;
        int rb  = n0 + row;
        int rac = ra < M ? ra : (M - 1);
        int rbc = rb < N ? rb : (N - 1);
        unsigned va = *(const unsigned*)(A  + (long long)rac * lda + kc);
        unsigned vb = *(const unsigned*)(Bm + (long long)rbc * ldb + kc);
        if (ra >= M || k >= K) va = 0;
        if (rb >= N || k >= K) vb = 0;
        *(unsigned*)(&As[row * LDSK + cp * 2]) = va;
        *(unsigned*)(&Bs[row * LDSK + cp * 2]) = vb;
      }
    }
    __syncthreads();

    #pragma unroll
    for (int ks = 0; ks < TK; ks += 32) {
      v16h a0 = frag_ld(&As[(wm +  0) * LDSK + ks], r16, kb);
      v16h a1 = frag_ld(&As[(wm + 16) * LDSK + ks], r16, kb);
      v16h b0 = frag_ld(&Bs[(wn +  0) * LDSK + ks], r16, kb);
      v16h b1 = frag_ld(&Bs[(wn + 16) * LDSK + ks], r16, kb);
      acc00 = wmma_f32(a0, b0, acc00);
      acc01 = wmma_f32(a0, b1, acc01);
      acc10 = wmma_f32(a1, b0, acc10);
      acc11 = wmma_f32(a1, b1, acc11);
    }

    __syncthreads();
  }

  // epilogue: C layout per ISA (VGPR v -> row v / v+8, lane -> col)
  const int cn  = lane & 15;
  const int rhi = (lane >> 4) * 8;
  #pragma unroll
  for (int i = 0; i < 2; ++i) {
    #pragma unroll
    for (int j = 0; j < 2; ++j) {
      v8f acc = (i == 0) ? (j == 0 ? acc00 : acc01) : (j == 0 ? acc10 : acc11);
      int rbase = m0 + wm + i * 16 + rhi;
      int col   = n0 + wn + j * 16 + cn;
      #pragma unroll
      for (int v = 0; v < 8; ++v) {
        int r = rbase + v;
        if (r < M && col < N) C[(long long)r * ldc + col] = acc[v];
      }
    }
  }
}

// ---------------------------------------------------------------------
// Scalar / elementwise kernels
// ---------------------------------------------------------------------
__global__ void k_f32_to_f16(const float* __restrict__ in, _Float16* __restrict__ out, long long n) {
  long long i = blockIdx.x * (long long)blockDim.x + threadIdx.x;
  if (i < n) out[i] = (_Float16)in[i];
}

// conv1: x (B,3,N) -> C1[32][B*N]
__global__ void k_conv1(const float* __restrict__ x, const float* __restrict__ w,
                        float* __restrict__ out, int B, int N) {
  long long cols = (long long)B * N;
  long long i = blockIdx.x * (long long)blockDim.x + threadIdx.x;
  if (i >= 32 * cols) return;
  int o = (int)(i / cols);
  long long col = i % cols;
  int b = (int)(col / N), l = (int)(col % N);
  const float* xp = x + ((long long)b * 3) * N + l;
  out[(long long)o * cols + col] =
      w[o*3+0]*xp[0] + w[o*3+1]*xp[N] + w[o*3+2]*xp[2*N];
}

// Generic per-channel stats: element(c,ib,ii) = x[ib*bStride + c*rowStride + ii*eStride] (+bias[c])
__global__ void k_stats(const float* __restrict__ x, const float* __restrict__ bias,
                        float* __restrict__ mean, float* __restrict__ rstd,
                        int nb, long long bStride, long long rowStride, long long eStride, long long cnt) {
  int c = blockIdx.x;
  long long total = (long long)nb * cnt;
  float s = 0.f, s2 = 0.f;
  for (long long i = threadIdx.x; i < total; i += blockDim.x) {
    long long ib = i / cnt, ii = i % cnt;
    float v = x[ib * bStride + (long long)c * rowStride + ii * eStride];
    if (bias) v += bias[c];
    s += v; s2 += v * v;
  }
  __shared__ float sh[256], sh2[256];
  sh[threadIdx.x] = s; sh2[threadIdx.x] = s2; __syncthreads();
  for (int st = 128; st > 0; st >>= 1) {
    if ((int)threadIdx.x < st) { sh[threadIdx.x] += sh[threadIdx.x + st]; sh2[threadIdx.x] += sh2[threadIdx.x + st]; }
    __syncthreads();
  }
  if (threadIdx.x == 0) {
    float m = sh[0] / (float)total;
    float v = sh2[0] / (float)total - m * m;
    mean[c] = m; rstd[c] = rsqrtf(v + 1e-5f);
  }
}

// bn+relu of (C x cols) f32 row-major -> transposed f16 [col][C] (ldo)
__global__ void k_bn_relu_T16(const float* __restrict__ x, const float* __restrict__ g, const float* __restrict__ b,
                              const float* __restrict__ mean, const float* __restrict__ rstd,
                              _Float16* __restrict__ out, int C, long long cols, int ldo) {
  long long i = blockIdx.x * (long long)blockDim.x + threadIdx.x;
  if (i >= (long long)C * cols) return;
  int c = (int)(i % C);
  long long col = i / C;
  float v = x[(long long)c * cols + col];
  v = g[c] * (v - mean[c]) * rstd[c] + b[c];
  v = v > 0.f ? v : 0.f;
  out[col * (long long)ldo + c] = (_Float16)v;
}

// same but f32 output
__global__ void k_bn_relu_Tf32(const float* __restrict__ x, const float* __restrict__ g, const float* __restrict__ b,
                               const float* __restrict__ mean, const float* __restrict__ rstd,
                               float* __restrict__ out, int C, long long cols, int ldo) {
  long long i = blockIdx.x * (long long)blockDim.x + threadIdx.x;
  if (i >= (long long)C * cols) return;
  int c = (int)(i % C);
  long long col = i / C;
  float v = x[(long long)c * cols + col];
  v = g[c] * (v - mean[c]) * rstd[c] + b[c];
  v = v > 0.f ? v : 0.f;
  out[col * (long long)ldo + c] = v;
}

// fused bn + relu + max over samples: C2 (C x B*G*S) -> xT[b][1+g][c]
__global__ void k_bn_relu_max(const float* __restrict__ x, const float* __restrict__ g, const float* __restrict__ bb,
                              const float* __restrict__ mean, const float* __restrict__ rstd,
                              float* __restrict__ outT, int C, int Bn, int G, int S, int nrows) {
  long long i = blockIdx.x * (long long)blockDim.x + threadIdx.x;
  if (i >= (long long)Bn * G * C) return;
  int c = (int)(i % C);
  long long bg = i / C;
  int b = (int)(bg / G), gi = (int)(bg % G);
  long long cols = (long long)Bn * G * S;
  const float* xp = x + (long long)c * cols + ((long long)(b * G + gi)) * S;
  float gm = g[c], bm = bb[c], mn = mean[c], rs = rstd[c];
  float m = 0.f;  // relu output >= 0
  for (int j = 0; j < S; ++j) {
    float v = gm * (xp[j] - mn) * rs + bm;
    v = v > 0.f ? v : 0.f;
    m = v > m ? v : m;
  }
  outT[((long long)(b * nrows + 1 + gi)) * C + c] = m;
}

__global__ void k_set_cls(const float* __restrict__ cls, float* __restrict__ xT, int B, int nrows, int C) {
  int i = blockIdx.x * blockDim.x + threadIdx.x;
  if (i >= B * C) return;
  int b = i / C, c = i % C;
  xT[((long long)b * nrows) * C + c] = cls[c];
}

__global__ void k_cls_mlp(const float* __restrict__ w, const float* __restrict__ xprevT,
                          float* __restrict__ xT, int B, int Co, int Ci, int nrPrev, int nrOut) {
  int i = blockIdx.x * blockDim.x + threadIdx.x;
  if (i >= B * Co) return;
  int b = i / Co, o = i % Co;
  const float* xp = xprevT + (long long)b * nrPrev * Ci;  // row 0
  float s = 0.f;
  for (int c = 0; c < Ci; ++c) s += w[o * Ci + c] * xp[c];
  xT[((long long)b * nrOut) * Co + o] = s;
}

__global__ void k_xyz_T(const float* __restrict__ x, float* __restrict__ xyz, int B, int N) {
  long long i = blockIdx.x * (long long)blockDim.x + threadIdx.x;
  if (i >= (long long)B * N * 3) return;
  int k = (int)(i % 3);
  long long bn = i / 3;
  int b = (int)(bn / N), n = (int)(bn % N);
  xyz[i] = x[((long long)b * 3 + k) * N + n];
}

__global__ void k_gather3(const float* __restrict__ xyz, const int* __restrict__ idx,
                          float* __restrict__ out, int B, int N, int S) {
  long long i = blockIdx.x * (long long)blockDim.x + threadIdx.x;
  if (i >= (long long)B * S * 3) return;
  int k = (int)(i % 3);
  long long bs = i / 3;
  int b = (int)(bs / S), s = (int)(bs % S);
  int j = idx[(long long)b * S + s];
  out[i] = xyz[((long long)b * N + j) * 3 + k];
}

// farthest point sampling: one block per batch (matches lax.scan semantics)
__global__ void k_fps(const float* __restrict__ xyz, int* __restrict__ out, int N, int np) {
  int b = blockIdx.x;
  const float* p = xyz + (long long)b * N * 3;
  __shared__ float dist[1024];
  __shared__ float bv[256];
  __shared__ int   bi[256];
  __shared__ int   sel;
  for (int j = threadIdx.x; j < N; j += blockDim.x) dist[j] = 1e10f;
  if (threadIdx.x == 0) sel = 0;
  __syncthreads();
  for (int it = 0; it < np; ++it) {
    int s = sel;
    if (threadIdx.x == 0) out[(long long)b * np + it] = s;
    float cx = p[s*3], cy = p[s*3+1], cz = p[s*3+2];
    float lbv = -1.f; int lbi = 0;
    for (int j = threadIdx.x; j < N; j += blockDim.x) {
      float dx = p[j*3]-cx, dy = p[j*3+1]-cy, dz = p[j*3+2]-cz;
      float d = dx*dx + dy*dy + dz*dz;
      float o = dist[j];
      d = d < o ? d : o;
      dist[j] = d;
      if (d > lbv) { lbv = d; lbi = j; }
    }
    bv[threadIdx.x] = lbv; bi[threadIdx.x] = lbi;
    __syncthreads();
    for (int st = 128; st > 0; st >>= 1) {
      if ((int)threadIdx.x < st) {
        float v2 = bv[threadIdx.x + st]; int i2 = bi[threadIdx.x + st];
        if (v2 > bv[threadIdx.x] || (v2 == bv[threadIdx.x] && i2 < bi[threadIdx.x])) {
          bv[threadIdx.x] = v2; bi[threadIdx.x] = i2;
        }
      }
      __syncthreads();
    }
    if (threadIdx.x == 0) sel = bi[0];
    __syncthreads();
  }
}

__global__ void k_ballquery(const float* __restrict__ xyz, const float* __restrict__ nxyz,
                            int* __restrict__ idx, int B, int N, int S, int ns, float r2) {
  int i = blockIdx.x * blockDim.x + threadIdx.x;
  if (i >= B * S) return;
  int b = i / S;
  const float* c = nxyz + (long long)i * 3;
  const float* p = xyz + (long long)b * N * 3;
  int* o = idx + (long long)i * ns;
  int cnt = 0;
  for (int j = 0; j < N && cnt < ns; ++j) {
    float dx = p[j*3]-c[0], dy = p[j*3+1]-c[1], dz = p[j*3+2]-c[2];
    if (dx*dx + dy*dy + dz*dz <= r2) o[cnt++] = j;
  }
  int f = cnt > 0 ? o[0] : 0;
  for (int k = cnt; k < ns; ++k) o[k] = f;
}

// grouping: F[col][0..d) = pts[idx]-center, F[col][d..2d) = center  (f16)
__global__ void k_group(const float* __restrict__ ptsT, const int* __restrict__ fpsIdx,
                        const int* __restrict__ idx, _Float16* __restrict__ F,
                        int B, int S, int ns, int d, int nrTot, int rowBase) {
  long long col = blockIdx.x * (long long)blockDim.x + threadIdx.x;
  long long cols = (long long)B * S * ns;
  if (col >= cols) return;
  long long bs = col / ns;
  int b = (int)(bs / S);
  int jp = idx[col];
  int jc = fpsIdx[bs];
  const float* pp = ptsT + ((long long)(b * nrTot + rowBase + jp)) * d;
  const float* pc = ptsT + ((long long)(b * nrTot + rowBase + jc)) * d;
  _Float16* o = F + col * (2LL * d);
  for (int c = 0; c < d; ++c) {
    float cv = pc[c];
    o[c]     = (_Float16)(pp[c] - cv);
    o[d + c] = (_Float16)cv;
  }
}

__global__ void k_softmax_row(float* __restrict__ S, int n) {
  long long r = blockIdx.x;
  float* row = S + r * (long long)n;
  __shared__ float sh[256];
  int t = threadIdx.x;
  float m = -1e30f;
  for (int j = t; j < n; j += 256) m = fmaxf(m, row[j]);
  sh[t] = m; __syncthreads();
  for (int st = 128; st > 0; st >>= 1) { if (t < st) sh[t] = fmaxf(sh[t], sh[t + st]); __syncthreads(); }
  m = sh[0]; __syncthreads();
  float s = 0.f;
  for (int j = t; j < n; j += 256) { float e = __expf(row[j] - m); row[j] = e; s += e; }
  sh[t] = s; __syncthreads();
  for (int st = 128; st > 0; st >>= 1) { if (t < st) sh[t] += sh[t + st]; __syncthreads(); }
  float inv = 1.f / sh[0];
  for (int j = t; j < n; j += 256) row[j] *= inv;
}

__global__ void k_colsum(const float* __restrict__ S, float* __restrict__ cs, int B, int n) {
  long long i = blockIdx.x * (long long)blockDim.x + threadIdx.x;
  if (i >= (long long)B * n) return;
  int b = (int)(i / n), m = (int)(i % n);
  const float* base = S + (long long)b * n * n + m;
  float s = 0.f;
  for (int j = 0; j < n; ++j) s += base[(long long)j * n];
  cs[i] = s;
}

// attT16[(b*n+m)][j] = S[b][j][m] / (1e-9 + colsum), zero-padded to ldt
__global__ void k_att_T16(const float* __restrict__ S, const float* __restrict__ cs,
                          _Float16* __restrict__ attT, int B, int n, int ldt) {
  long long i = blockIdx.x * (long long)blockDim.x + threadIdx.x;
  if (i >= (long long)B * n) return;
  int b = (int)(i / n), m = (int)(i % n);
  float inv = 1.f / (1e-9f + cs[i]);
  const float* src = S + (long long)b * n * n + m;
  _Float16* o = attT + i * (long long)ldt;
  int j = 0;
  for (; j < n; ++j) o[j] = (_Float16)(src[(long long)j * n] * inv);
  for (; j < ldt; ++j) o[j] = (_Float16)0.f;
}

// V f32 (b,C,n) + bias[c] -> f16 rows padded to ldo
__global__ void k_bias_f16(const float* __restrict__ x, const float* __restrict__ bias,
                           _Float16* __restrict__ out, int B, int C, int n, int ldo) {
  long long i = blockIdx.x * (long long)blockDim.x + threadIdx.x;
  if (i >= (long long)B * C) return;
  int c = (int)(i % C);
  const float* src = x + i * (long long)n;
  _Float16* o = out + i * (long long)ldo;
  float bv = bias[c];
  int j = 0;
  for (; j < n; ++j) o[j] = (_Float16)(src[j] + bv);
  for (; j < ldo; ++j) o[j] = (_Float16)0.f;
}

// D16[(b,n,c)] = xT - xr(b,c,n)
__global__ void k_sub_T16(const float* __restrict__ xT, const float* __restrict__ xr,
                          _Float16* __restrict__ out, int B, int n, int C) {
  long long i = blockIdx.x * (long long)blockDim.x + threadIdx.x;
  if (i >= (long long)B * n * C) return;
  int c = (int)(i % C);
  long long bn = i / C;
  int b = (int)(bn / n), j = (int)(bn % n);
  out[i] = (_Float16)(xT[i] - xr[((long long)b * C + c) * n + j]);
}

// xT += relu(bn(T + t_b))  (residual)
__global__ void k_sa_out(const float* __restrict__ T, const float* __restrict__ tb,
                         const float* __restrict__ g, const float* __restrict__ bb,
                         const float* __restrict__ mean, const float* __restrict__ rstd,
                         float* __restrict__ xT, int B, int n, int C) {
  long long i = blockIdx.x * (long long)blockDim.x + threadIdx.x;
  if (i >= (long long)B * n * C) return;
  int c = (int)(i % C);
  long long bn = i / C;
  int b = (int)(bn / n), j = (int)(bn % n);
  float t = T[((long long)b * C + c) * n + j] + tb[c];
  float v = g[c] * (t - mean[c]) * rstd[c] + bb[c];
  v = v > 0.f ? v : 0.f;
  xT[i] += v;
}

__global__ void k_row0_f16(const float* __restrict__ xT, _Float16* __restrict__ H, int B, int nrows, int C) {
  int i = blockIdx.x * blockDim.x + threadIdx.x;
  if (i >= B * C) return;
  int b = i / C, c = i % C;
  H[i] = (_Float16)xT[((long long)b * nrows) * C + c];
}

__global__ void k_bnc_lrelu_f16(const float* __restrict__ X, const float* __restrict__ bias,
                                const float* __restrict__ g, const float* __restrict__ bb,
                                const float* __restrict__ mean, const float* __restrict__ rstd,
                                _Float16* __restrict__ out, int Bn, int F) {
  int i = blockIdx.x * blockDim.x + threadIdx.x;
  if (i >= Bn * F) return;
  int f = i % F;
  float v = X[i] + (bias ? bias[f] : 0.f);
  v = g[f] * (v - mean[f]) * rstd[f] + bb[f];
  v = v > 0.f ? v : 0.2f * v;
  out[i] = (_Float16)v;
}

__global__ void k_add_bias(float* __restrict__ X, const float* __restrict__ bias, int Bn, int F) {
  int i = blockIdx.x * blockDim.x + threadIdx.x;
  if (i >= Bn * F) return;
  X[i] += bias[i % F];
}

// ---------------------------------------------------------------------
// Host side
// ---------------------------------------------------------------------
static void gemm(hipStream_t st, const _Float16* A, const _Float16* B, float* C,
                 int M, int N, int K, int lda, int ldb, int ldc,
                 long long sA, long long sB, long long sC, int batch) {
  dim3 g((unsigned)((N + TN - 1) / TN), (unsigned)((M + TM - 1) / TM), (unsigned)batch);
  k_gemm_nt<<<g, GEMM_THREADS, 0, st>>>(A, B, C, M, N, K, lda, ldb, ldc, sA, sB, sC);
}

extern "C" void kernel_launch(void* const* d_in, const int* in_sizes, int n_in,
                              void* d_out, int out_size, void* d_ws, size_t ws_size,
                              hipStream_t stream) {
  (void)in_sizes; (void)n_in; (void)out_size; (void)ws_size;
  const int B = 32, N0 = 1024;

  auto F32p = [&](int i) { return (const float*)d_in[i]; };
  const float* x        = F32p(0);
  const float* conv1_w  = F32p(1);
  const float* conv2_w  = F32p(2);
  const float* bn1_g    = F32p(3);
  const float* bn1_b    = F32p(4);
  const float* bn2_g    = F32p(5);
  const float* bn2_b    = F32p(6);
  const float* cls_tok  = F32p(7);
  const float *lo_c1[4], *lo_c2[4], *lo_g1[4], *lo_b1[4], *lo_g2[4], *lo_b2[4];
  for (int i = 0; i < 4; ++i) {
    int base = 8 + i * 6;
    lo_c1[i] = F32p(base); lo_c2[i] = F32p(base+1);
    lo_g1[i] = F32p(base+2); lo_b1[i] = F32p(base+3);
    lo_g2[i] = F32p(base+4); lo_b2[i] = F32p(base+5);
  }
  const float* mlp01 = F32p(32);
  const float* mlp12 = F32p(33);
  const float* mlp23 = F32p(34);
  const float *sa_qk[4], *sa_vw[4], *sa_vb[4], *sa_tw[4], *sa_tb[4], *sa_g[4], *sa_b[4];
  for (int i = 0; i < 4; ++i) {
    int base = 35 + i * 7;
    sa_qk[i] = F32p(base);   sa_vw[i] = F32p(base+1); sa_vb[i] = F32p(base+2);
    sa_tw[i] = F32p(base+3); sa_tb[i] = F32p(base+4);
    sa_g[i]  = F32p(base+5); sa_b[i]  = F32p(base+6);
  }
  const float* lin1  = F32p(63);
  const float* bn6g  = F32p(64);
  const float* bn6b  = F32p(65);
  const float* lin2w = F32p(66);
  const float* lin2b = F32p(67);
  const float* bn7g  = F32p(68);
  const float* bn7b  = F32p(69);
  const float* lin3w = F32p(70);
  const float* lin3b = F32p(71);

  // ---- workspace bump allocator (deterministic) ----
  char* wsp = (char*)d_ws;
  size_t off = 0;
  auto alloc = [&](size_t bytes) -> void* {
    off = (off + 255) & ~(size_t)255;
    void* p = wsp + off;
    off += bytes;
    return p;
  };

  // big reusable GEMM buffers
  float*    Cbuf = (float*)alloc((size_t)64 * 524288 * 4);   // 134 MB
  _Float16* F16A = (_Float16*)alloc((size_t)64 * 524288 * 2); // 67 MB (grouped feats / G0)
  _Float16* F16B = (_Float16*)alloc((size_t)64 * 524288 * 2); // 67 MB (bn-relu f16)

  // persistent activations
  float* f_ptsT = (float*)alloc((size_t)B * N0 * 32 * 4);
  float* xyz0   = (float*)alloc((size_t)B * N0 * 3 * 4);
  float* nx1    = (float*)alloc((size_t)B * 512 * 3 * 4);
  float* nx2    = (float*)alloc((size_t)B * 256 * 3 * 4);
  float* nx3    = (float*)alloc((size_t)B * 128 * 3 * 4);
  float* nx4    = (float*)alloc((size_t)B * 64  * 3 * 4);
  float* x0T    = (float*)alloc((size_t)B * 513 * 64  * 4);
  float* x1T    = (float*)alloc((size_t)B * 257 * 128 * 4);
  float* x2T    = (float*)alloc((size_t)B * 129 * 256 * 4);
  float* x3T    = (float*)alloc((size_t)B * 65  * 512 * 4);
  int* fpsIdx   = (int*)alloc((size_t)B * 512 * 4);
  int* ballIdx  = (int*)alloc((size_t)B * 512 * 32 * 4);
  float* meanB  = (float*)alloc(512 * 4);
  float* rstdB  = (float*)alloc(512 * 4);

  // SA scratch (sized for max over the 4 layers)
  const size_t nC_max  = 33280;   // max n*C
  const size_t nq_max  = 8320;    // max n*q
  const size_t nn_max  = 263169;  // max n*n (513^2)
  const size_t nnp_max = 266760;  // max n*npad (513*520)
  const size_t Cnp_max = 36864;   // max C*npad
  _Float16* saX16  = (_Float16*)alloc((size_t)B * nC_max * 2);
  float*    saQf   = (float*)alloc((size_t)B * nq_max * 4);
  _Float16* saQ16  = (_Float16*)alloc((size_t)B * nq_max * 2);
  float*    saS    = (float*)alloc((size_t)B * nn_max * 4);
  float*    saCs   = (float*)alloc((size_t)B * 513 * 4);
  _Float16* saAttT = (_Float16*)alloc((size_t)B * nnp_max * 2);
  float*    saVf   = (float*)alloc((size_t)B * nC_max * 4);
  _Float16* saV16  = (_Float16*)alloc((size_t)B * Cnp_max * 2);
  float*    saXR   = (float*)alloc((size_t)B * nC_max * 4);
  _Float16* saD16  = (_Float16*)alloc((size_t)B * nC_max * 2);
  float*    saTf   = (float*)alloc((size_t)B * nC_max * 4);

  // head scratch
  _Float16* H16   = (_Float16*)alloc((size_t)B * 512 * 2);
  float*    H1f   = (float*)alloc((size_t)B * 256 * 4);
  _Float16* H1_16 = (_Float16*)alloc((size_t)B * 256 * 2);
  float*    H2f   = (float*)alloc((size_t)B * 128 * 4);
  _Float16* H2_16 = (_Float16*)alloc((size_t)B * 128 * 2);

  // f16 weights
  auto cvt = [&](const float* src, long long n) -> _Float16* {
    _Float16* d = (_Float16*)alloc((size_t)n * 2);
    k_f32_to_f16<<<nblk(n), 256, 0, stream>>>(src, d, n);
    return d;
  };
  _Float16* w_conv2 = cvt(conv2_w, 32 * 32);
  _Float16 *w_lo1[4], *w_lo2[4];
  const int loC[4] = {64, 128, 256, 512};
  for (int i = 0; i < 4; ++i) {
    w_lo1[i] = cvt(lo_c1[i], (long long)loC[i] * loC[i]);
    w_lo2[i] = cvt(lo_c2[i], (long long)loC[i] * loC[i]);
  }
  _Float16 *w_qk[4], *w_vw[4], *w_tw[4];
  const int saC[4] = {64, 128, 256, 512};
  for (int i = 0; i < 4; ++i) {
    w_qk[i] = cvt(sa_qk[i], (long long)(saC[i] / 4) * saC[i]);
    w_vw[i] = cvt(sa_vw[i], (long long)saC[i] * saC[i]);
    w_tw[i] = cvt(sa_tw[i], (long long)saC[i] * saC[i]);
  }
  _Float16* w_lin1 = cvt(lin1, 256 * 512);
  _Float16* w_lin2 = cvt(lin2w, 128 * 256);
  _Float16* w_lin3 = cvt(lin3w, 40 * 128);

  // ---------------- SA layer ----------------
  auto run_sa = [&](float* xT, int n, int C, int li) {
    int q = C / 4;
    int npad = (n + 7) & ~7;
    long long tot = (long long)B * n * C;
    k_f32_to_f16<<<nblk(tot), 256, 0, stream>>>(xT, saX16, tot);
    // Q[n,q] = X * qk^T  (per batch)
    gemm(stream, saX16, w_qk[li], saQf, n, q, C, C, C, q,
         (long long)n * C, 0, (long long)n * q, B);
    k_f32_to_f16<<<nblk((long long)B * n * q), 256, 0, stream>>>(saQf, saQ16, (long long)B * n * q);
    // scores S = Q Q^T
    gemm(stream, saQ16, saQ16, saS, n, n, q, q, q, n,
         (long long)n * q, (long long)n * q, (long long)n * n, B);
    k_softmax_row<<<(unsigned)(B * n), 256, 0, stream>>>(saS, n);
    k_colsum<<<nblk((long long)B * n), 256, 0, stream>>>(saS, saCs, B, n);
    k_att_T16<<<nblk((long long)B * n), 256, 0, stream>>>(saS, saCs, saAttT, B, n, npad);
    // V = v_w X^T (+bias -> f16 padded)
    gemm(stream, w_vw[li], saX16, saVf, C, n, C, C, C, n,
         0, (long long)n * C, (long long)C * n, B);
    k_bias_f16<<<nblk((long long)B * C), 256, 0, stream>>>(saVf, sa_vb[li], saV16, B, C, n, npad);
    // XR = V attT^T
    gemm(stream, saV16, saAttT, saXR, C, n, n, npad, npad, n,
         (long long)C * npad, (long long)n * npad, (long long)C * n, B);
    k_sub_T16<<<nblk(tot), 256, 0, stream>>>(xT, saXR, saD16, B, n, C);
    // T = t_w (x - xr)
    gemm(stream, w_tw[li], saD16, saTf, C, n, C, C, C, n,
         0, (long long)n * C, (long long)C * n, B);
    k_stats<<<dim3((unsigned)C), 256, 0, stream>>>(saTf, sa_tb[li], meanB, rstdB,
        B, (long long)C * n, (long long)n, 1LL, (long long)n);
    k_sa_out<<<nblk(tot), 256, 0, stream>>>(saTf, sa_tb[li], sa_g[li], sa_b[li],
        meanB, rstdB, xT, B, n, C);
  };

  // ---------------- sample+group+local_op stage ----------------
  auto run_stage = [&](int Nsrc, int np, float radius, int ns,
                       const float* xyzIn, float* newXyz,
                       const float* ptsT, int nrTot, int rowBase, int d,
                       int li, float* outXT) {
    int co = loC[li];
    int ci = 2 * d;
    k_fps<<<dim3((unsigned)B), 256, 0, stream>>>(xyzIn, fpsIdx, Nsrc, np);
    k_gather3<<<nblk((long long)B * np * 3), 256, 0, stream>>>(xyzIn, fpsIdx, newXyz, B, Nsrc, np);
    k_ballquery<<<nblk((long long)B * np), 256, 0, stream>>>(xyzIn, newXyz, ballIdx,
        B, Nsrc, np, ns, radius * radius);
    long long cols = (long long)B * np * ns;
    k_group<<<nblk(cols), 256, 0, stream>>>(ptsT, fpsIdx, ballIdx, F16A, B, np, ns, d, nrTot, rowBase);
    // conv1 of local op
    gemm(stream, w_lo1[li], F16A, Cbuf, co, (int)cols, ci, ci, ci, (int)cols, 0, 0, 0, 1);
    k_stats<<<dim3((unsigned)co), 256, 0, stream>>>(Cbuf, nullptr, meanB, rstdB, 1, 0, cols, 1, cols);
    k_bn_relu_T16<<<nblk((long long)co * cols), 256, 0, stream>>>(Cbuf, lo_g1[li], lo_b1[li],
        meanB, rstdB, F16B, co, cols, co);
    // conv2 of local op
    gemm(stream, w_lo2[li], F16B, Cbuf, co, (int)cols, co, co, co, (int)cols, 0, 0, 0, 1);
    k_stats<<<dim3((unsigned)co), 256, 0, stream>>>(Cbuf, nullptr, meanB, rstdB, 1, 0, cols, 1, cols);
    k_bn_relu_max<<<nblk((long long)B * np * co), 256, 0, stream>>>(Cbuf, lo_g2[li], lo_b2[li],
        meanB, rstdB, outXT, co, B, np, ns, np + 1);
  };

  // ================= pipeline =================
  long long cols0 = (long long)B * N0;

  // input conv1 -> bn -> relu -> f16
  k_conv1<<<nblk(32 * cols0), 256, 0, stream>>>(x, conv1_w, Cbuf, B, N0);
  k_stats<<<dim3(32), 256, 0, stream>>>(Cbuf, nullptr, meanB, rstdB, 1, 0, cols0, 1, cols0);
  k_bn_relu_T16<<<nblk(32 * cols0), 256, 0, stream>>>(Cbuf, bn1_g, bn1_b, meanB, rstdB, F16A, 32, cols0, 32);
  // conv2 (WMMA) -> bn -> relu -> (b,n,32) f32
  gemm(stream, w_conv2, F16A, Cbuf, 32, (int)cols0, 32, 32, 32, (int)cols0, 0, 0, 0, 1);
  k_stats<<<dim3(32), 256, 0, stream>>>(Cbuf, nullptr, meanB, rstdB, 1, 0, cols0, 1, cols0);
  k_bn_relu_Tf32<<<nblk(32 * cols0), 256, 0, stream>>>(Cbuf, bn2_g, bn2_b, meanB, rstdB, f_ptsT, 32, cols0, 32);
  k_xyz_T<<<nblk((long long)B * N0 * 3), 256, 0, stream>>>(x, xyz0, B, N0);

  // stage 0: np=512, r=0.15, ns=32, d=32 -> x0T (b,513,64)
  run_stage(N0, 512, 0.15f, 32, xyz0, nx1, f_ptsT, N0, 0, 32, 0, x0T);
  k_set_cls<<<nblk((long long)B * 64), 256, 0, stream>>>(cls_tok, x0T, B, 513, 64);
  run_sa(x0T, 513, 64, 0);

  // stage 1: np=256, r=0.2, ns=32, pts = x0T rows 1.. (d=64) -> x1T (b,257,128)
  run_stage(512, 256, 0.2f, 32, nx1, nx2, x0T, 513, 1, 64, 1, x1T);
  k_cls_mlp<<<nblk((long long)B * 128), 256, 0, stream>>>(mlp01, x0T, x1T, B, 128, 64, 513, 257);
  run_sa(x1T, 257, 128, 1);

  // stage 2: np=128, r=0.25, ns=24, d=128 -> x2T (b,129,256)
  run_stage(256, 128, 0.25f, 24, nx2, nx3, x1T, 257, 1, 128, 2, x2T);
  k_cls_mlp<<<nblk((long long)B * 256), 256, 0, stream>>>(mlp12, x1T, x2T, B, 256, 128, 257, 129);
  run_sa(x2T, 129, 256, 2);

  // stage 3: np=64, r=0.3, ns=16, d=256 -> x3T (b,65,512)
  run_stage(128, 64, 0.3f, 16, nx3, nx4, x2T, 129, 1, 256, 3, x3T);
  k_cls_mlp<<<nblk((long long)B * 512), 256, 0, stream>>>(mlp23, x2T, x3T, B, 512, 256, 129, 65);
  run_sa(x3T, 65, 512, 3);

  // head: cls row -> lin1 -> bn6 -> lrelu -> lin2 -> bn7 -> lrelu -> lin3 (+bias)
  k_row0_f16<<<nblk((long long)B * 512), 256, 0, stream>>>(x3T, H16, B, 65, 512);
  gemm(stream, H16, w_lin1, H1f, B, 256, 512, 512, 512, 256, 0, 0, 0, 1);
  k_stats<<<dim3(256), 256, 0, stream>>>(H1f, nullptr, meanB, rstdB, 1, 0, 1, 256, B);
  k_bnc_lrelu_f16<<<nblk((long long)B * 256), 256, 0, stream>>>(H1f, nullptr, bn6g, bn6b,
      meanB, rstdB, H1_16, B, 256);
  gemm(stream, H1_16, w_lin2, H2f, B, 128, 256, 256, 256, 128, 0, 0, 0, 1);
  k_stats<<<dim3(128), 256, 0, stream>>>(H2f, lin2b, meanB, rstdB, 1, 0, 1, 128, B);
  k_bnc_lrelu_f16<<<nblk((long long)B * 128), 256, 0, stream>>>(H2f, lin2b, bn7g, bn7b,
      meanB, rstdB, H2_16, B, 128);
  gemm(stream, H2_16, w_lin3, (float*)d_out, B, 40, 128, 128, 128, 40, 0, 0, 0, 1);
  k_add_bias<<<nblk((long long)B * 40), 256, 0, stream>>>((float*)d_out, lin3b, B, 40);
}
